// HadamardTransform_43722767073461
// MI455X (gfx1250) — matrix operand order repaired and checked by reading, not compiled, not run on hardware
//
#include <hip/hip_runtime.h>

// CDNA5 / gfx1250 Hadamard block rotation via bf16x3-split WMMA.
// y = (x . H) * scale, H = +-1 exactly representable in bf16,
// x split into 3 exact bf16 terms (hi/mid/lo truncation splits),
// processed one split at a time to keep VGPR usage <= 256.

typedef __attribute__((ext_vector_type(16))) __bf16          v16bf;
typedef __attribute__((ext_vector_type(16))) unsigned short  v16s;
typedef __attribute__((ext_vector_type(8)))  float           v8f;
typedef __attribute__((ext_vector_type(4)))  float           f32x4;

#define THREADS 256
#define WAVES_PER_BLOCK 8   // wave32

__global__ __launch_bounds__(THREADS)
__attribute__((amdgpu_waves_per_eu(4)))
void hadamard_wmma_bf16x3_kernel(const float* __restrict__ x,
                                 const float* __restrict__ W,
                                 float* __restrict__ out,
                                 int nStrips)
{
    // B fragments: 32 frags (kc 0..3 x nt 0..7) x 32 lanes x 16 bf16 = 32KB.
    // Fragment-major so each lane reads one contiguous 32B v16bf operand.
    __shared__ alignas(32) unsigned short ldsB[16384];

    const int tid = threadIdx.x;

    // ---- Stage sign(H) as bf16 +-1 in WMMA B-operand layout ----
    for (int g = tid; g < 16384; g += THREADS) {
        const int f   = g >> 9;        // fragment index = kc*8 + nt
        const int rem = g & 511;
        const int l   = rem >> 4;      // lane within fragment
        const int e   = rem & 15;      // element within v16bf
        const int hh  = l >> 4;
        const int kc  = f >> 3;
        const int nt  = f & 7;
        const int K   = kc * 32 + hh * 16 + e;   // B row (K dim)
        const int N   = nt * 16 + (l & 15);      // B col (N dim)
        const unsigned int wbits = __builtin_bit_cast(unsigned int, W[K * 128 + N]);
        ldsB[g] = (wbits & 0x80000000u) ? 0xBF80u : 0x3F80u;   // -1.0bf16 : +1.0bf16
    }
    __syncthreads();

    const float scale = W[0];          // +1/sqrt(128) as stored in W (f32)
    const int lane = tid & 31;
    const int m    = lane & 15;        // A: M index ; D: N index within tile
    const int h    = lane >> 4;        // lane half selects K sub-range / M+8
    const int gwave  = blockIdx.x * WAVES_PER_BLOCK + (tid >> 5);
    const int wstep  = gridDim.x * WAVES_PER_BLOCK;

    for (int strip = gwave; strip < nStrips; strip += wstep) {
        const int rowBase = strip << 4;                     // 16 rows per strip
        const float* px = x + (size_t)(rowBase + m) * 128;

        // Prefetch next strip's rows (2 instrs x 32 lanes cover all 64 lines)
        if (strip + wstep < nStrips) {
            const float* pn = x + (size_t)(((strip + wstep) << 4) + m) * 128;
            __builtin_prefetch(pn + h * 32, 0, 3);
            __builtin_prefetch(pn + 64 + h * 32, 0, 3);
        }

        v8f acc[8] = {};

        #pragma unroll
        for (int kc = 0; kc < 4; ++kc) {
            const int k0 = kc * 32;

            // B fragments for this K-chunk stay in registers (8 x v16bf)
            v16bf bfrag[8];
            #pragma unroll
            for (int nt = 0; nt < 8; ++nt) {
                bfrag[nt] = __builtin_bit_cast(
                    v16bf,
                    *reinterpret_cast<const v16s*>(&ldsB[((kc * 8 + nt) * 32 + lane) * 16]));
            }

            // A 16x32 bf16 layout: lane half h holds K in [16h,16h+16)
            float rr[16];
            {
                const f32x4 t0 = *(const f32x4*)(px + k0 + 16 * h);
                const f32x4 t1 = *(const f32x4*)(px + k0 + 16 * h + 4);
                const f32x4 t2 = *(const f32x4*)(px + k0 + 16 * h + 8);
                const f32x4 t3 = *(const f32x4*)(px + k0 + 16 * h + 12);
                #pragma unroll
                for (int j = 0; j < 4; ++j) {
                    rr[j]      = t0[j];
                    rr[4 + j]  = t1[j];
                    rr[8 + j]  = t2[j];
                    rr[12 + j] = t3[j];
                }
            }

            // Process splits sequentially: only one v16bf A fragment live at a time.
            #pragma unroll
            for (int s = 0; s < 3; ++s) {
                v16s cs;
                #pragma unroll
                for (int e = 0; e < 16; ++e) {
                    const unsigned int u = __builtin_bit_cast(unsigned int, rr[e]);
                    cs[e] = (unsigned short)(u >> 16);
                    if (s < 2) {   // residual for the next split (exact in f32)
                        rr[e] -= __builtin_bit_cast(float, u & 0xFFFF0000u);
                    }
                }
                const v16bf a = __builtin_bit_cast(v16bf, cs);

                #pragma unroll
                for (int nt = 0; nt < 8; ++nt) {
                    // (neg_a, A, neg_b, B, c_mod, C, reuse_a, reuse_b)
                    acc[nt] = __builtin_amdgcn_wmma_f32_16x16x32_bf16(
                        false, a, false, bfrag[nt], (short)0, acc[nt], false, false);
                }
            }
        }

        // D 16x16 f32 layout: VGPR v -> row (v + 8h), col = nt*16 + m
        float* pout = out + (size_t)rowBase * 128;
        #pragma unroll
        for (int nt = 0; nt < 8; ++nt) {
            #pragma unroll
            for (int v = 0; v < 8; ++v) {
                pout[(size_t)(v + 8 * h) * 128 + nt * 16 + m] = acc[nt][v] * scale;
            }
        }
    }
}

extern "C" void kernel_launch(void* const* d_in, const int* in_sizes, int n_in,
                              void* d_out, int out_size, void* d_ws, size_t ws_size,
                              hipStream_t stream) {
    const float* x = (const float*)d_in[0];
    const float* W = (const float*)d_in[1];
    float* out = (float*)d_out;

    const int nStrips = out_size / (16 * 128);              // 16 rows x 128 cols per strip
    int blocks = (nStrips + WAVES_PER_BLOCK * 4 - 1) / (WAVES_PER_BLOCK * 4); // ~4 strips/wave
    if (blocks > 1024) blocks = 1024;
    if (blocks < 1) blocks = 1;

    hadamard_wmma_bf16x3_kernel<<<blocks, THREADS, 0, stream>>>(x, W, out, nStrips);
}
